// QLinear_63934883169111
// MI455X (gfx1250) — 8-bit matrix operand order repaired and checked by reading, compiled, not run on hardware
//
#include <hip/hip_runtime.h>
#include <stdint.h>

typedef int   v8i    __attribute__((ext_vector_type(8)));
typedef float v4f    __attribute__((ext_vector_type(4)));
typedef int   v4i    __attribute__((ext_vector_type(4)));
typedef int   v4i_vs __attribute__((vector_size(16)));

#define AS1 __attribute__((address_space(1)))
#define AS3 __attribute__((address_space(3)))

#define TOKENS 8192
#define KDIM   4096
#define NDIM   4096
#define TM     128
#define TN     128
#define TK     128
#define NKT    (KDIM / TK)

#if defined(__has_builtin)
#  if __has_builtin(__builtin_amdgcn_global_load_async_to_lds_b128)
#    define USE_ASYNC_LDS 1
#  endif
#endif
#ifndef USE_ASYNC_LDS
#  define USE_ASYNC_LDS 0
#endif

#if USE_ASYNC_LDS
#  if __has_builtin(__builtin_amdgcn_s_wait_asynccnt)
#    define WAIT_ASYNC(n) __builtin_amdgcn_s_wait_asynccnt(n)
#  else
#    define WAIT_ASYNC(n) asm volatile("s_wait_asynccnt %0" ::"i"(n) : "memory")
#  endif
#endif

__global__ __launch_bounds__(256) void pack_w_int8(const int* __restrict__ src,
                                                   int* __restrict__ dst) {
    const int i = blockIdx.x * 256 + threadIdx.x;
    v4i v = __builtin_nontemporal_load((const v4i*)src + i);
    dst[i] = (v.x & 0xFF) | ((v.y & 0xFF) << 8) | ((v.z & 0xFF) << 16) | (v.w << 24);
}

__global__ __launch_bounds__(256) void quant_x_int8(const float* __restrict__ x,
                                                    int* __restrict__ xq,
                                                    float* __restrict__ iscale) {
    __shared__ float red[256];
    const int t   = threadIdx.x;
    const int row = blockIdx.x;
    const v4f* xr = (const v4f*)(x + (size_t)row * KDIM);

    v4f v[4];
    float m = 0.f;
#pragma unroll
    for (int c = 0; c < 4; ++c) {
        v[c] = __builtin_nontemporal_load(&xr[c * 256 + t]);
        m = fmaxf(m, fmaxf(fmaxf(fabsf(v[c].x), fabsf(v[c].y)),
                           fmaxf(fabsf(v[c].z), fabsf(v[c].w))));
    }
    red[t] = m;
    __syncthreads();
#pragma unroll
    for (int s = 128; s > 0; s >>= 1) {
        if (t < s) red[t] = fmaxf(red[t], red[t + s]);
        __syncthreads();
    }
    const float amax = red[0];
    const float inv  = (amax > 0.f) ? (127.0f / amax) : 0.f;
    if (t == 0) iscale[row] = amax * (1.0f / 127.0f);

    int* xqr = xq + (size_t)row * (KDIM / 4);
#pragma unroll
    for (int c = 0; c < 4; ++c) {
        int q0 = min(127, max(-127, __float2int_rn(v[c].x * inv)));
        int q1 = min(127, max(-127, __float2int_rn(v[c].y * inv)));
        int q2 = min(127, max(-127, __float2int_rn(v[c].z * inv)));
        int q3 = min(127, max(-127, __float2int_rn(v[c].w * inv)));
        xqr[c * 256 + t] = (q0 & 0xFF) | ((q1 & 0xFF) << 8) | ((q2 & 0xFF) << 16) | (q3 << 24);
    }
}

__device__ __forceinline__ void stage_tile(const unsigned char* __restrict__ gA,
                                           const unsigned char* __restrict__ gB,
                                           unsigned char* lA, unsigned char* lB, int t) {
#pragma unroll
    for (int c = 0; c < 4; ++c) {
        const int lin = c * 256 + t;
        const int row = lin >> 3;
        const int kq  = (lin & 7) << 4;
#if USE_ASYNC_LDS
        __builtin_amdgcn_global_load_async_to_lds_b128(
            (AS1 v4i_vs*)(gA + (size_t)row * KDIM + kq),
            (AS3 v4i_vs*)(lA + row * TK + kq), 0, 0);
        __builtin_amdgcn_global_load_async_to_lds_b128(
            (AS1 v4i_vs*)(gB + (size_t)row * KDIM + kq),
            (AS3 v4i_vs*)(lB + row * TK + kq), 0, 0);
#else
        *(uint4*)(lA + row * TK + kq) = *(const uint4*)(gA + (size_t)row * KDIM + kq);
        *(uint4*)(lB + row * TK + kq) = *(const uint4*)(gB + (size_t)row * KDIM + kq);
#endif
    }
}

__global__ __launch_bounds__(256) void qgemm_wmma(const unsigned char* __restrict__ xq,
                                                  const unsigned char* __restrict__ wq,
                                                  const float* __restrict__ iscale,
                                                  const float* __restrict__ wscale,
                                                  const float* __restrict__ bias,
                                                  float* __restrict__ out) {
    __shared__ __align__(16) unsigned char As[2][TM * TK];
    __shared__ __align__(16) unsigned char Bs[2][TN * TK];

    const int t     = threadIdx.x;
    const int lane  = t & 31;
    const int wave  = t >> 5;
    const int waveM = wave >> 2;
    const int waveN = wave & 3;
    const int mBase = blockIdx.y * TM;
    const int nBase = blockIdx.x * TN;

    const int halfK = lane >> 4;
    const int l16   = lane & 15;

    const unsigned char* gA = xq + (size_t)mBase * KDIM;
    const unsigned char* gB = wq + (size_t)nBase * KDIM;

    v8i acc[4][2];
#pragma unroll
    for (int i = 0; i < 4; ++i)
#pragma unroll
        for (int j = 0; j < 2; ++j)
#pragma unroll
            for (int e = 0; e < 8; ++e) acc[i][j][e] = 0;

    stage_tile(gA, gB, As[0], Bs[0], t);

#pragma unroll 1
    for (int kt = 0; kt < NKT; ++kt) {
        const int cur = kt & 1;

        if (kt + 1 < NKT) {
            stage_tile(gA + (size_t)(kt + 1) * TK, gB + (size_t)(kt + 1) * TK,
                       As[1 - cur], Bs[1 - cur], t);
#if USE_ASYNC_LDS
            WAIT_ASYNC(8);
#endif
        } else {
#if USE_ASYNC_LDS
            WAIT_ASYNC(0);
#endif
        }
        __syncthreads();

#pragma unroll
        for (int h = 0; h < 2; ++h) {
            v8i bf[2];
#pragma unroll
            for (int j = 0; j < 2; ++j) {
                const unsigned char* bp =
                    Bs[cur] + (waveN * 32 + j * 16 + l16) * TK + h * 64 + halfK * 16;
                const uint4 lo = *(const uint4*)(bp);
                const uint4 hi = *(const uint4*)(bp + 32);
                bf[j][0] = (int)lo.x; bf[j][1] = (int)lo.y; bf[j][2] = (int)lo.z; bf[j][3] = (int)lo.w;
                bf[j][4] = (int)hi.x; bf[j][5] = (int)hi.y; bf[j][6] = (int)hi.z; bf[j][7] = (int)hi.w;
            }
            v8i af[4];
#pragma unroll
            for (int i = 0; i < 4; ++i) {
                const unsigned char* ap =
                    As[cur] + (waveM * 64 + i * 16 + l16) * TK + h * 64 + halfK * 16;
#pragma unroll
                for (int jj = 0; jj < 4; ++jj) {
                    const uint2 d = *(const uint2*)(ap + (jj >> 1) * 32 + (jj & 1) * 8);
                    af[i][2 * jj]     = (int)d.x;
                    af[i][2 * jj + 1] = (int)d.y;
                }
            }
#pragma unroll
            for (int i = 0; i < 4; ++i)
#pragma unroll
                for (int j = 0; j < 2; ++j)
                    acc[i][j] = __builtin_amdgcn_wmma_i32_16x16x64_iu8(
                        true, af[i], true, bf[j], acc[i][j], false, false);
        }
        __syncthreads();
    }

#pragma unroll
    for (int i = 0; i < 4; ++i) {
#pragma unroll
        for (int j = 0; j < 2; ++j) {
            const int n = nBase + waveN * 32 + j * 16 + l16;
            const float wsn = wscale[n];
            const float bn  = bias[n];
#pragma unroll
            for (int r = 0; r < 8; ++r) {
                const int m = mBase + waveM * 64 + i * 16 + halfK * 8 + r;
                const float val = (float)acc[i][j][r] * iscale[m] * wsn + bn;
                __builtin_nontemporal_store(val, &out[(size_t)m * NDIM + n]);
            }
        }
    }
}

extern "C" void kernel_launch(void* const* d_in, const int* in_sizes, int n_in,
                              void* d_out, int out_size, void* d_ws, size_t ws_size,
                              hipStream_t stream) {
    const float* x      = (const float*)d_in[0];
    const int*   wq32   = (const int*)d_in[1];
    const float* wscale = (const float*)d_in[2];
    const float* bias   = (const float*)d_in[3];
    float*       out    = (float*)d_out;

    char* ws = (char*)d_ws;
    unsigned char* xq8 = (unsigned char*)ws;
    unsigned char* wq8 = xq8 + (size_t)TOKENS * KDIM;
    float*      iscale = (float*)(wq8 + (size_t)NDIM * KDIM);

    pack_w_int8<<<(NDIM * KDIM / 4) / 256, 256, 0, stream>>>(wq32, (int*)wq8);

    quant_x_int8<<<TOKENS, 256, 0, stream>>>(x, (int*)xq8, iscale);

    dim3 grid(NDIM / TN, TOKENS / TM);
    qgemm_wmma<<<grid, 256, 0, stream>>>(xq8, wq8, iscale, wscale, bias, out);
}
